// FLAMELayer_51161650430260
// MI455X (gfx1250) — matrix operand order repaired and checked by reading, not compiled, not run on hardware
//
#include <hip/hip_runtime.h>
#include <hip/hip_bf16.h>

// ---------------- problem dimensions ----------------
#define VV    5023            // vertices
#define NV    15069           // V*3 output columns
#define NPAD  15072           // 314 vertex-tiles * 48
#define K400  400
#define KP    448             // 436 (400 shape + 36 pose) padded to 14*32
#define BN    2048

typedef __attribute__((ext_vector_type(8)))  float  v8f;
typedef __attribute__((ext_vector_type(8)))  __bf16 v8bf;
typedef __attribute__((ext_vector_type(16))) __bf16 v16bf;
typedef __attribute__((ext_vector_type(4)))  unsigned int u32x4;
typedef __attribute__((ext_vector_type(8)))  int i32x8;
typedef __attribute__((ext_vector_type(4)))  int i32x4;

static __device__ __forceinline__ __bf16 f2bf(float f) {
  union { float f; unsigned u; } in; in.f = f;
  unsigned u = in.u;
  u += 0x7FFFu + ((u >> 16) & 1u);               // round-to-nearest-even
  union { unsigned short s; __bf16 b; } out;
  out.s = (unsigned short)(u >> 16);
  return out.b;
}

// ---------------------------------------------------------------------------
// K1: pack dirs_ext = [shapedirs | posedirs^T | 0-pad] -> bf16 [NPAD][KP]
// ---------------------------------------------------------------------------
__global__ __launch_bounds__(256) void prep_dirs(
    const float* __restrict__ shapedirs,   // [V,3,400] == [NV][400]
    const float* __restrict__ posedirs,    // [36][NV]
    __bf16* __restrict__ dirs_bf)          // [NPAD][KP]
{
  int g = blockIdx.x * blockDim.x + threadIdx.x;
  if (g >= NPAD * (KP / 8)) return;
  int n  = g / (KP / 8);
  int k0 = (g % (KP / 8)) * 8;
  v8bf o;
  if (n < NV && k0 < K400) {
    const float4* p = (const float4*)(shapedirs + (size_t)n * K400 + k0);
    float4 a = p[0], b = p[1];
    o[0]=f2bf(a.x); o[1]=f2bf(a.y); o[2]=f2bf(a.z); o[3]=f2bf(a.w);
    o[4]=f2bf(b.x); o[5]=f2bf(b.y); o[6]=f2bf(b.z); o[7]=f2bf(b.w);
  } else {
#pragma unroll
    for (int e = 0; e < 8; ++e) {
      int k = k0 + e;
      float v = 0.f;
      if (n < NV && k >= K400 && k < 436) v = posedirs[(size_t)(k - K400) * NV + n];
      o[e] = f2bf(v);
    }
  }
  *(v8bf*)(dirs_bf + (size_t)n * KP + k0) = o;
}

// ---------------------------------------------------------------------------
// K2: JS[15][400] = J_regressor @ shapedirs ; J0[15] = J_regressor @ v_template
// ---------------------------------------------------------------------------
__global__ __launch_bounds__(128) void prep_JS(
    const float* __restrict__ Jreg,        // [5][V]
    const float* __restrict__ shapedirs,   // [NV][400]
    const float* __restrict__ vtempl,      // [NV]
    float* __restrict__ JS, float* __restrict__ J0)
{
  int t = blockIdx.x * blockDim.x + threadIdx.x;
  if (t < 6000) {
    int jc = t / K400, l = t % K400;
    int j = jc / 3, c = jc % 3;
    const float* jr = Jreg + (size_t)j * VV;
    float acc = 0.f;
    for (int v = 0; v < VV; ++v)
      acc += jr[v] * shapedirs[(size_t)(v * 3 + c) * K400 + l];
    JS[t] = acc;
  } else if (t < 6015) {
    int jc = t - 6000, j = jc / 3, c = jc % 3;
    const float* jr = Jreg + (size_t)j * VV;
    float acc = 0.f;
    for (int v = 0; v < VV; ++v) acc += jr[v] * vtempl[v * 3 + c];
    J0[jc] = acc;
  }
}

// ---------------------------------------------------------------------------
// K3: per-batch (one wave each): Rodrigues, joints, FK chain -> A_rel[b][5][12],
//     pose feature, and bf16 pack of betas_ext[b][KP]
// ---------------------------------------------------------------------------
__global__ __launch_bounds__(32) void prep_batch(
    const float* __restrict__ shape, const float* __restrict__ expr,
    const float* __restrict__ rot,   const float* __restrict__ neck,
    const float* __restrict__ jaw,   const float* __restrict__ eyes,
    const float* __restrict__ JS,    const float* __restrict__ J0,
    __bf16* __restrict__ betas_bf,   float* __restrict__ A_out)
{
  int b = blockIdx.x;
  int lane = threadIdx.x;
  __shared__ float sR[5][9];
  __shared__ float sJ[5][3];
  __shared__ float sPF[36];

  if (lane < 5) {                       // Rodrigues for joint `lane`
    float r0, r1, r2;
    if      (lane == 0) { r0 = rot [b*3+0]; r1 = rot [b*3+1]; r2 = rot [b*3+2]; }
    else if (lane == 1) { r0 = neck[b*3+0]; r1 = neck[b*3+1]; r2 = neck[b*3+2]; }
    else if (lane == 2) { r0 = jaw [b*3+0]; r1 = jaw [b*3+1]; r2 = jaw [b*3+2]; }
    else if (lane == 3) { r0 = eyes[b*6+0]; r1 = eyes[b*6+1]; r2 = eyes[b*6+2]; }
    else                { r0 = eyes[b*6+3]; r1 = eyes[b*6+4]; r2 = eyes[b*6+5]; }
    float ang = sqrtf(r0*r0 + r1*r1 + r2*r2) + 1e-8f;
    float x = r0/ang, y = r1/ang, z = r2/ang;
    float s = sinf(ang), c = cosf(ang), oc = 1.f - c;
    sR[lane][0] = 1.f - oc*(y*y + z*z);
    sR[lane][1] = -s*z + oc*x*y;
    sR[lane][2] =  s*y + oc*x*z;
    sR[lane][3] =  s*z + oc*x*y;
    sR[lane][4] = 1.f - oc*(x*x + z*z);
    sR[lane][5] = -s*x + oc*y*z;
    sR[lane][6] = -s*y + oc*x*z;
    sR[lane][7] =  s*x + oc*y*z;
    sR[lane][8] = 1.f - oc*(x*x + y*y);
  }
  if (lane < 15) {                      // J[b] = J0 + JS @ betas
    const float* js = JS + (size_t)lane * K400;
    const float* sh = shape + (size_t)b * 300;
    const float* ex = expr  + (size_t)b * 100;
    float acc = J0[lane];
    for (int l = 0; l < 300; ++l) acc += sh[l] * js[l];
    for (int l = 0; l < 100; ++l) acc += ex[l] * js[300 + l];
    sJ[lane / 3][lane % 3] = acc;
  }
  __syncthreads();

  for (int i = lane; i < 36; i += 32) { // pose feature = (R[1:] - I).flatten
    int j = 1 + i / 9, e = i % 9;
    sPF[i] = sR[j][e] - ((e == 0 || e == 4 || e == 8) ? 1.f : 0.f);
  }
  __syncthreads();

  for (int k = lane; k < KP; k += 32) { // bf16 pack of betas_ext row
    float v = (k < 300) ? shape[(size_t)b*300 + k]
            : (k < K400) ? expr[(size_t)b*100 + (k - 300)]
            : (k < 436) ? sPF[k - K400] : 0.f;
    betas_bf[(size_t)b * KP + k] = f2bf(v);
  }

  if (lane == 0) {                      // FK chain + rest-pose removal
    float A[5][12];
#pragma unroll
    for (int m = 0; m < 3; ++m) {
      A[0][m*4+0] = sR[0][m*3+0]; A[0][m*4+1] = sR[0][m*3+1];
      A[0][m*4+2] = sR[0][m*3+2]; A[0][m*4+3] = sJ[0][m];
    }
    for (int j = 1; j < 5; ++j) {
      int p = (j == 1) ? 0 : 1;         // PARENTS = [-1,0,1,1,1]
      float t0 = sJ[j][0]-sJ[p][0], t1 = sJ[j][1]-sJ[p][1], t2 = sJ[j][2]-sJ[p][2];
#pragma unroll
      for (int m = 0; m < 3; ++m) {
        float a0 = A[p][m*4+0], a1 = A[p][m*4+1], a2 = A[p][m*4+2];
#pragma unroll
        for (int n = 0; n < 3; ++n)
          A[j][m*4+n] = a0*sR[j][0*3+n] + a1*sR[j][1*3+n] + a2*sR[j][2*3+n];
        A[j][m*4+3] = a0*t0 + a1*t1 + a2*t2 + A[p][m*4+3];
      }
    }
    float* out = A_out + (size_t)b * 60;
    for (int j = 0; j < 5; ++j)
#pragma unroll
      for (int m = 0; m < 3; ++m) {
        float tc = A[j][m*4+0]*sJ[j][0] + A[j][m*4+1]*sJ[j][1] + A[j][m*4+2]*sJ[j][2];
        out[j*12 + m*4 + 0] = A[j][m*4+0];
        out[j*12 + m*4 + 1] = A[j][m*4+1];
        out[j*12 + m*4 + 2] = A[j][m*4+2];
        out[j*12 + m*4 + 3] = A[j][m*4+3] - tc;
      }
  }
}

// ---------------------------------------------------------------------------
// K4: 4-wave workgroup. TDM loads the shared 48-col x 448-K bf16 B-tile into
// LDS once; each wave runs a 16x48 WMMA tile (A from global, B via ds_load)
// and a fused LBS epilogue. B L2-traffic drops 4x vs one-wave tiles.
// ---------------------------------------------------------------------------
static __device__ __forceinline__ v16bf cat8(v8bf lo, v8bf hi) {
  return __builtin_shufflevector(lo, hi, 0,1,2,3,4,5,6,7,8,9,10,11,12,13,14,15);
}

__global__ __launch_bounds__(128) void gemm_lbs(
    const __bf16* __restrict__ betas_bf,   // [BN][KP]
    const __bf16* __restrict__ dirs_bf,    // [NPAD][KP]
    const float*  __restrict__ vtempl,     // [NV]
    const float*  __restrict__ A_rel,      // [BN][5][12]
    const float*  __restrict__ lbsw,       // [V][5]
    float* __restrict__ out)               // [BN][NV]
{
  int tid   = threadIdx.x;
  int lane  = tid & 31;
  int w     = tid >> 5;                    // wave id 0..3
  int vbase = blockIdx.x * 16;             // first vertex of tile
  int n0    = blockIdx.x * 48;             // first output column
  int bbase = blockIdx.y * 64;             // first batch row of workgroup
  int brow0 = bbase + w * 16;              // this wave's first batch row

  __shared__ __bf16 smB[48 * KP];          // 43008 B shared B-tile
  __shared__ float  smC[4][16][48];
  __shared__ float  smO[4][16][48];
  __shared__ float  smA[4][16][60];
  __shared__ float  smW[16][5];

  // per-wave A_rel preload (read in epilogue by same wave: wave-ordered LDS)
  for (int i = lane; i < 960; i += 32)
    smA[w][i / 60][i % 60] = A_rel[(size_t)(brow0 + i / 60) * 60 + i % 60];
  if (tid < 80) {                          // shared skinning weights
    int v = vbase + tid / 5;
    smW[tid / 5][tid % 5] = (v < VV) ? lbsw[(size_t)v * 5 + tid % 5] : 0.f;
  }

  // ---- Tensor Data Mover: global [48 rows x 448 bf16] -> LDS, one shot ----
  if (w == 0) {
    unsigned lds_off = (unsigned)(uintptr_t)(&smB[0]);
    unsigned long long ga =
        (unsigned long long)(uintptr_t)(dirs_bf + (size_t)n0 * KP);
    u32x4 g0;
    g0[0] = 1u;                                            // count=1 valid D#
    g0[1] = lds_off;                                       // lds_addr
    g0[2] = (unsigned)(ga & 0xFFFFFFFFull);                // global_addr lo
    g0[3] = (unsigned)((ga >> 32) & 0x1FFFFFFull) | (2u << 30); // hi + type=2
    i32x8 g1;
    g1[0] = (int)(1u << 16);          // data_size=2B, no multicast/pad/iterate
    g1[1] = (int)((unsigned)KP << 16);        // tensor_dim0 = 448
    g1[2] = (int)(48u << 16);                 // tensor_dim1 = 48
    g1[3] = (int)((unsigned)KP << 16);        // tile_dim0 = 448
    g1[4] = 48;                               // tile_dim1 = 48
    g1[5] = KP;                               // tensor_dim0_stride = 448
    g1[6] = 0;
    g1[7] = 0;
    i32x4 gz = {0, 0, 0, 0};
#if defined(__clang_major__) && (__clang_major__ >= 23)
    i32x8 gz8 = {0, 0, 0, 0, 0, 0, 0, 0};
    __builtin_amdgcn_tensor_load_to_lds(g0, g1, gz, gz, gz8, 0);
#else
    __builtin_amdgcn_tensor_load_to_lds(g0, g1, gz, gz, 0);
#endif
    __builtin_amdgcn_s_wait_tensorcnt(0);
  }
  __syncthreads();                        // B-tile + smW visible to all waves

  int half = lane >> 4;                    // 0: lanes 0-15, 1: lanes 16-31
  int mrow = lane & 15;
  const __bf16* aPtr = betas_bf + (size_t)(brow0 + mrow) * KP + half * 16;
  const __bf16* bl0  = smB + (size_t)mrow        * KP + half * 16;
  const __bf16* bl1  = smB + (size_t)(16 + mrow) * KP + half * 16;
  const __bf16* bl2  = smB + (size_t)(32 + mrow) * KP + half * 16;

  v8f c0 = {}, c1 = {}, c2 = {};
#pragma unroll
  for (int kk = 0; kk < KP / 32; ++kk) {
    int ko = kk * 32;
    v16bf a  = cat8(*(const v8bf*)(aPtr + ko), *(const v8bf*)(aPtr + ko + 8));
    v16bf b0 = cat8(*(const v8bf*)(bl0 + ko),  *(const v8bf*)(bl0 + ko + 8));
    v16bf b1 = cat8(*(const v8bf*)(bl1 + ko),  *(const v8bf*)(bl1 + ko + 8));
    v16bf b2 = cat8(*(const v8bf*)(bl2 + ko),  *(const v8bf*)(bl2 + ko + 8));
    c0 = __builtin_amdgcn_wmma_f32_16x16x32_bf16(false, a, false, b0, (short)0, c0, false, false);
    c1 = __builtin_amdgcn_wmma_f32_16x16x32_bf16(false, a, false, b1, (short)0, c1, false, false);
    c2 = __builtin_amdgcn_wmma_f32_16x16x32_bf16(false, a, false, b2, (short)0, c2, false, false);
  }

  // C -> LDS with v_template added (this is v_posed for the tile)
#pragma unroll
  for (int i = 0; i < 8; ++i) {
    int row = half * 8 + i;
    int n;
    n = n0 + mrow;      smC[w][row][mrow]      = c0[i] + ((n < NV) ? vtempl[n] : 0.f);
    n = n0 + 16 + mrow; smC[w][row][16 + mrow] = c1[i] + ((n < NV) ? vtempl[n] : 0.f);
    n = n0 + 32 + mrow; smC[w][row][32 + mrow] = c2[i] + ((n < NV) ? vtempl[n] : 0.f);
  }

  // fused LBS (all LDS data wave-local or barrier-covered)
  for (int it = 0; it < 8; ++it) {
    int idx  = it * 32 + lane;
    int row  = idx >> 4;
    int vloc = idx & 15;
    float vp0 = smC[w][row][vloc * 3 + 0];
    float vp1 = smC[w][row][vloc * 3 + 1];
    float vp2 = smC[w][row][vloc * 3 + 2];
    const float* Ab = smA[w][row];
    float w0 = smW[vloc][0], w1 = smW[vloc][1], w2 = smW[vloc][2],
          w3 = smW[vloc][3], w4 = smW[vloc][4];
#pragma unroll
    for (int m = 0; m < 3; ++m) {
      float t0 = w0*Ab[0*12+m*4+0] + w1*Ab[1*12+m*4+0] + w2*Ab[2*12+m*4+0]
               + w3*Ab[3*12+m*4+0] + w4*Ab[4*12+m*4+0];
      float t1 = w0*Ab[0*12+m*4+1] + w1*Ab[1*12+m*4+1] + w2*Ab[2*12+m*4+1]
               + w3*Ab[3*12+m*4+1] + w4*Ab[4*12+m*4+1];
      float t2 = w0*Ab[0*12+m*4+2] + w1*Ab[1*12+m*4+2] + w2*Ab[2*12+m*4+2]
               + w3*Ab[3*12+m*4+2] + w4*Ab[4*12+m*4+2];
      float t3 = w0*Ab[0*12+m*4+3] + w1*Ab[1*12+m*4+3] + w2*Ab[2*12+m*4+3]
               + w3*Ab[3*12+m*4+3] + w4*Ab[4*12+m*4+3];
      smO[w][row][vloc * 3 + m] = t0*vp0 + t1*vp1 + t2*vp2 + t3;
    }
  }

  // coalesced store (per-wave rows)
  for (int i = lane; i < 16 * 48; i += 32) {
    int row = i / 48, col = i % 48, n = n0 + col;
    if (n < NV) out[(size_t)(brow0 + row) * NV + n] = smO[w][row][col];
  }
}

// ---------------------------------------------------------------------------
extern "C" void kernel_launch(void* const* d_in, const int* in_sizes, int n_in,
                              void* d_out, int out_size, void* d_ws, size_t ws_size,
                              hipStream_t stream)
{
  const float* shape  = (const float*)d_in[0];
  const float* expr   = (const float*)d_in[1];
  const float* rot    = (const float*)d_in[2];
  const float* neck   = (const float*)d_in[3];
  const float* jaw    = (const float*)d_in[4];
  const float* eyes   = (const float*)d_in[5];
  const float* vtempl = (const float*)d_in[6];
  const float* shdirs = (const float*)d_in[7];
  const float* psdirs = (const float*)d_in[8];
  const float* jreg   = (const float*)d_in[9];
  const float* lbsw   = (const float*)d_in[10];
  float* out = (float*)d_out;

  char* ws = (char*)d_ws;
  __bf16* dirs_bf  = (__bf16*)ws;                                   // 13,504,512 B
  __bf16* betas_bf = (__bf16*)(ws + (size_t)NPAD * KP * 2);         //  1,835,008 B
  float*  JS       = (float*)(ws + (size_t)NPAD * KP * 2 + (size_t)BN * KP * 2);
  float*  J0       = JS + 6016;
  float*  Arel     = J0 + 64;                                       // BN*60 floats

  prep_dirs<<<(NPAD * (KP / 8) + 255) / 256, 256, 0, stream>>>(shdirs, psdirs, dirs_bf);
  prep_JS<<<(6016 + 127) / 128, 128, 0, stream>>>(jreg, shdirs, vtempl, JS, J0);
  prep_batch<<<BN, 32, 0, stream>>>(shape, expr, rot, neck, jaw, eyes,
                                    JS, J0, betas_bf, Arel);
  gemm_lbs<<<dim3(NPAD / 48, BN / 64), 128, 0, stream>>>(betas_bf, dirs_bf, vtempl,
                                                         Arel, lbsw, out);
}
